// TemporalSelfAttention_3839700763041
// MI455X (gfx1250) — matrix operand order repaired and checked by reading, not compiled, not run on hardware
//
#include <hip/hip_runtime.h>
#include <hip/hip_bf16.h>
#include <cmath>

typedef __attribute__((ext_vector_type(16))) _Float16 v16h;
typedef __attribute__((ext_vector_type(8)))  _Float16 v8h;
typedef __attribute__((ext_vector_type(8)))  float    v8f;

#define B_  2
#define P_  24
#define N_  207
#define D_  64
#define H_  8
#define DK_ 8
#define DH_ 128
constexpr int TOK = B_ * P_ * N_;   // 9936 tokens
constexpr int C3  = 3 * H_ * DK_;   // 192 QKV channels
constexpr int KCH = 258;            // 256 chunks of h(x)x outer product + 2 bias chunks (K = 8256)
constexpr int NT  = C3 / 16;        // 12 N-tiles
constexpr int MT  = 3;              // M-tiles (of 16 tokens) per workgroup -> 48 tokens
constexpr int NWG = TOK / (16 * MT);// 207 workgroups
constexpr float EPS_ = 1e-5f;

// ---------------------------------------------------------------------------
// Kernel 0: pack W2 (+ folded b2 rows) into WMMA-B fragment-ready f16 layout.
// Fragment for (ntile, kchunk): lane L holds 16 halves = rows K=(L/16)*16+0..15,
// col N = L%16 (contiguous k order, same as the A fragments), stored contiguously.
// ---------------------------------------------------------------------------
__global__ void pack_w2_kernel(const float* __restrict__ W2,
                               const float* __restrict__ b2,
                               _Float16* __restrict__ w2p) {
    int bid   = blockIdx.x;
    int ntile = bid / KCH;
    int kk    = bid % KCH;
    int L     = threadIdx.x;              // 0..31
    int c     = ntile * 16 + (L & 15);    // output channel (s,h,k) flattened
    v16h frag;
#pragma unroll
    for (int e = 0; e < 16; ++e) {
        int Klocal = ((L >> 4) << 4) + e;         // grp*16 + e
        int kg     = kk * 32 + Klocal;            // global K index
        float val;
        if (kg < DH_ * D_) {                      // h (x) x region: k = j*64 + d
            int j = kg >> 6, d = kg & 63;
            val = W2[(size_t)(c * D_ + d) * DH_ + j];
        } else {                                  // bias rows: G = x[d], B = b2
            int d = kg - DH_ * D_;
            val = b2[c * D_ + d];
        }
        frag[e] = (_Float16)val;
    }
    v16h* dst = (v16h*)(w2p + (size_t)(ntile * KCH + kk) * 512);
    dst[L] = frag;                                // 32B aligned per lane
}

// ---------------------------------------------------------------------------
// Kernel 1: h = relu(c_x @ W1^T + b1) -> f16 ; also cache x in f16.
// One block (128 threads) per token.
// ---------------------------------------------------------------------------
__global__ void h_mlp_kernel(const float* __restrict__ x,
                             const float* __restrict__ cx,
                             const float* __restrict__ W1,
                             const float* __restrict__ b1,
                             _Float16* __restrict__ h16,
                             _Float16* __restrict__ x16) {
    int t = blockIdx.x;
    int j = threadIdx.x;                  // 0..127
    __shared__ float sc[D_];
    if (j < D_) sc[j] = cx[(size_t)t * D_ + j];
    __syncthreads();
    float acc = b1[j];
#pragma unroll 8
    for (int d = 0; d < D_; ++d) acc = fmaf(W1[j * D_ + d], sc[d], acc);
    acc = fmaxf(acc, 0.f);
    h16[(size_t)t * DH_ + j] = (_Float16)acc;
    if (j < D_) x16[(size_t)t * D_ + j] = (_Float16)x[(size_t)t * D_ + j];
}

// ---------------------------------------------------------------------------
// Kernel 2: big GEMM QKV = G @ W2p, G[t,(j,d)] = h[t,j]*x[t,d], bias folded
// into the 2 tail K-chunks. 48 tokens (3 M-tiles) per workgroup, 4 waves x
// 3 N-tiles, 9 accumulators per wave: each B fragment load feeds 3 WMMAs.
// A fragments built on the fly (two contiguous 8-half x runs scaled by h).
// ---------------------------------------------------------------------------
__global__ void __launch_bounds__(128)
qkv_wmma_kernel(const _Float16* __restrict__ h16,
                const _Float16* __restrict__ x16,
                const _Float16* __restrict__ w2p,
                float* __restrict__ qkv) {
    __shared__ __align__(16) _Float16 sh_h[16 * MT * DH_];  // 12 KB
    __shared__ __align__(16) _Float16 sh_x[16 * MT * D_];   //  6 KB
    int tid = threadIdx.x;
    int t0  = blockIdx.x * (16 * MT);
    for (int i = tid; i < 16 * MT * DH_; i += 128) sh_h[i] = h16[(size_t)t0 * DH_ + i];
    for (int i = tid; i < 16 * MT * D_;  i += 128) sh_x[i] = x16[(size_t)t0 * D_  + i];
    __syncthreads();

    int wv   = tid >> 5;        // wave 0..3 -> N-tiles wv*3 .. wv*3+2
    int lane = tid & 31;
    int Ml   = lane & 15;       // A-operand row within an M-tile
    int grp  = lane >> 4;       // lane group selects K half

    v8f acc[MT][3] = {};

    for (int kk = 0; kk < KCH; ++kk) {
        bool bias  = (kk >= 256);
        int  j     = kk >> 1;
        int  dbase = bias ? (kk - 256) * 32 : (kk & 1) * 32;

        // Build MT A fragments (contiguous k order, same as W2p:
        // grp 0 -> K 0..15, grp 1 -> K 16..31; contiguous 8-half runs)
        v16h a[MT];
#pragma unroll
        for (int m = 0; m < MT; ++m) {
            int row = m * 16 + Ml;
            _Float16 hv = bias ? (_Float16)1.0f : sh_h[row * DH_ + j];
            const v8h* xr = (const v8h*)(sh_x + row * D_);
            v8h x0 = xr[(dbase >> 3) + 2 * grp];       // K = dbase + grp*16 + 0..7
            v8h x1 = xr[(dbase >> 3) + 2 * grp + 1];   // K = dbase + grp*16 + 8..15
#pragma unroll
            for (int e = 0; e < 8; ++e) { a[m][e] = hv * x0[e]; a[m][8 + e] = hv * x1[e]; }
        }

#pragma unroll
        for (int i = 0; i < 3; ++i) {
            int ntile = wv * 3 + i;
            const _Float16* bbase = w2p + (size_t)(ntile * KCH + kk) * 512;
            if (i == 0 && kk + 8 < KCH)   // gfx1250 global_prefetch_b8, 8 chunks ahead
                __builtin_prefetch((const void*)(bbase + 8 * 512 + lane * 16), 0, 0);
            v16h b = ((const v16h*)bbase)[lane];
#pragma unroll
            for (int m = 0; m < MT; ++m)
                acc[m][i] = __builtin_amdgcn_wmma_f32_16x16x32_f16(
                    false, a[m], false, b, (short)0, acc[m][i], false, false);
        }
    }

    // C layout: N = lane%16, M = grp*8 + v
#pragma unroll
    for (int m = 0; m < MT; ++m) {
#pragma unroll
        for (int i = 0; i < 3; ++i) {
            int col = (wv * 3 + i) * 16 + Ml;
#pragma unroll
            for (int v = 0; v < 8; ++v) {
                int row = m * 16 + grp * 8 + v;
                qkv[(size_t)(t0 + row) * C3 + col] = acc[m][i][v];
            }
        }
    }
}

// ---------------------------------------------------------------------------
// Kernel 3: tiny causal attention per (b, n, h): P=24, DK=8. One wave each.
// ---------------------------------------------------------------------------
__global__ void attn_kernel(const float* __restrict__ qkv,
                            float* __restrict__ attn) {
    int bid = blockIdx.x;                 // b*N*H problems
    int b   = bid / (N_ * H_);
    int rem = bid % (N_ * H_);
    int n   = rem / H_;
    int h   = rem % H_;
    __shared__ float Km[P_][DK_];
    __shared__ float Vm[P_][DK_];
    int L = threadIdx.x;
    for (int idx = L; idx < P_ * DK_; idx += 32) {
        int p = idx >> 3, k = idx & 7;
        size_t t = (size_t)(b * P_ + p) * N_ + n;
        Km[p][k] = qkv[t * C3 + 64  + h * DK_ + k];
        Vm[p][k] = qkv[t * C3 + 128 + h * DK_ + k];
    }
    __syncthreads();
    if (L < P_) {
        int p = L;
        size_t tq = (size_t)(b * P_ + p) * N_ + n;
        float q[DK_];
#pragma unroll
        for (int k = 0; k < DK_; ++k) q[k] = qkv[tq * C3 + h * DK_ + k];
        const float scale = 0.35355339059327373f;   // 1/sqrt(8)
        float s[P_];
        float m = -INFINITY;
        for (int qq = 0; qq <= p; ++qq) {           // causal: only qq <= p
            float d = 0.f;
#pragma unroll
            for (int k = 0; k < DK_; ++k) d = fmaf(q[k], Km[qq][k], d);
            d *= scale;
            s[qq] = d;
            m = fmaxf(m, d);
        }
        float denom = 0.f;
        for (int qq = 0; qq <= p; ++qq) { s[qq] = __expf(s[qq] - m); denom += s[qq]; }
        float inv = 1.f / denom;
        float o[DK_] = {};
        for (int qq = 0; qq <= p; ++qq) {
            float a = s[qq] * inv;
#pragma unroll
            for (int k = 0; k < DK_; ++k) o[k] = fmaf(a, Vm[qq][k], o[k]);
        }
#pragma unroll
        for (int k = 0; k < DK_; ++k) attn[tq * (H_ * DK_) + h * DK_ + k] = o[k];
    }
}

// ---------------------------------------------------------------------------
// Kernel 4: out-proj + residual + LayerNorm. One block (64 threads) per token.
// ---------------------------------------------------------------------------
__global__ void out_ln_kernel(const float* __restrict__ attn,
                              const float* __restrict__ x,
                              const float* __restrict__ Wout,
                              const float* __restrict__ gamma,
                              const float* __restrict__ beta,
                              float* __restrict__ out) {
    int t = blockIdx.x;
    int d = threadIdx.x;                  // 0..63
    __shared__ float sa[D_];
    __shared__ float red[D_];
    sa[d] = attn[(size_t)t * D_ + d];
    __syncthreads();
    float acc = x[(size_t)t * D_ + d];    // residual
#pragma unroll 8
    for (int i = 0; i < D_; ++i) acc = fmaf(Wout[d * D_ + i], sa[i], acc);
    red[d] = acc;
    __syncthreads();
    float sum = 0.f, sq = 0.f;
    for (int i = 0; i < D_; ++i) { float v = red[i]; sum += v; sq += v * v; }
    float mu  = sum * (1.f / D_);
    float var = sq * (1.f / D_) - mu * mu;
    out[(size_t)t * D_ + d] = (acc - mu) * rsqrtf(var + EPS_) * gamma[d] + beta[d];
}

// ---------------------------------------------------------------------------
extern "C" void kernel_launch(void* const* d_in, const int* in_sizes, int n_in,
                              void* d_out, int out_size, void* d_ws, size_t ws_size,
                              hipStream_t stream) {
    const float* inputs   = (const float*)d_in[0];
    const float* c_inputs = (const float*)d_in[1];
    const float* W1       = (const float*)d_in[2];
    const float* b1       = (const float*)d_in[3];
    const float* W2       = (const float*)d_in[4];
    const float* b2       = (const float*)d_in[5];
    const float* Wout     = (const float*)d_in[6];
    const float* gamma    = (const float*)d_in[7];
    const float* beta     = (const float*)d_in[8];
    float* out = (float*)d_out;

    char*  ws  = (char*)d_ws;
    size_t off = 0;
    auto alloc = [&](size_t bytes) -> void* {
        void* p = ws + off;
        off += (bytes + 255) & ~(size_t)255;
        return p;
    };
    _Float16* h16  = (_Float16*)alloc((size_t)TOK * DH_ * 2);        // 2.5 MB
    _Float16* x16  = (_Float16*)alloc((size_t)TOK * D_  * 2);        // 1.3 MB
    _Float16* w2p  = (_Float16*)alloc((size_t)NT * KCH * 512 * 2);   // 3.2 MB
    float*    qkv  = (float*)   alloc((size_t)TOK * C3 * 4);         // 7.6 MB
    float*    attn = (float*)   alloc((size_t)TOK * D_ * 4);         // 2.5 MB

    pack_w2_kernel <<<NT * KCH,    32, 0, stream>>>(W2, b2, w2p);
    h_mlp_kernel   <<<TOK,        128, 0, stream>>>(inputs, c_inputs, W1, b1, h16, x16);
    qkv_wmma_kernel<<<NWG,        128, 0, stream>>>(h16, x16, w2p, qkv);
    attn_kernel    <<<B_ * N_ * H_, 32, 0, stream>>>(qkv, attn);
    out_ln_kernel  <<<TOK,         64, 0, stream>>>(attn, inputs, Wout, gamma, beta, out);
}
